// SyntaxesAttention_67654324847030
// MI455X (gfx1250) — matrix operand order repaired and checked by reading, not compiled, not run on hardware
//
#include <hip/hip_runtime.h>
#include <stdint.h>

// Problem constants (fixed by the reference setup)
#define BB    4
#define SS    4096
#define DD    2048
#define HH    16
#define DHh   128
#define RR    4
#define WIN_  128
#define CSc   1024            // SS / RR
#define WSs   3968            // SS - WIN_
#define NEGV  (-1000000000.0f)

typedef __attribute__((ext_vector_type(16))) __bf16 v16bf;
typedef __attribute__((ext_vector_type(8)))  __bf16 v8bf;
typedef __attribute__((ext_vector_type(8)))  float  v8f;
typedef __attribute__((ext_vector_type(4)))  float  v4f;

union Frag { v16bf v; v8bf h[2]; };

__device__ __forceinline__ v8f wmma_bf16(const Frag& a, const Frag& b, v8f c) {
  return __builtin_amdgcn_wmma_f32_16x16x32_bf16(false, a.v, false, b.v,
                                                 (short)0, c, false, false);
}

// CDNA5 async copy: HBM -> LDS, 16 bytes per lane, tracked by ASYNCcnt.
// Generic pointers to __shared__ carry the LDS byte offset in their low 32
// bits (ISA 10.2), so we can pass them directly.
__device__ __forceinline__ void async_copy_b128(const void* gptr, void* lptr) {
  uint32_t lds  = (uint32_t)(uintptr_t)lptr;
  uint64_t addr = (uint64_t)(uintptr_t)gptr;
  asm volatile("global_load_async_to_lds_b128 %0, %1, off"
               :: "v"(lds), "v"(addr) : "memory");
}
__device__ __forceinline__ void async_wait0() {
  asm volatile("s_wait_asynccnt 0x0" ::: "memory");
}

// ---------------------------------------------------------------------------
// Prep: chunk-mean of inputs over R=4 rows -> bf16  (exploits linearity of Wq)
// ---------------------------------------------------------------------------
__global__ __launch_bounds__(256) void prep_mean_k(const float* __restrict__ x,
                                                   __bf16* __restrict__ xc) {
  size_t i  = (size_t)blockIdx.x * 256 + threadIdx.x;    // over BB*CSc*DD
  size_t d  = i & (DD - 1);
  size_t bc = i >> 11;
  size_t c  = bc & (CSc - 1);
  size_t b  = bc >> 10;
  const float* p = x + ((b * SS + c * RR) << 11) + d;
  float s = p[0] + p[DD] + p[2 * DD] + p[3 * DD];
  xc[i] = (__bf16)(s * 0.25f);
}

// Window slice -> bf16
__global__ __launch_bounds__(256) void prep_win_k(const float* __restrict__ x,
                                                  __bf16* __restrict__ wx) {
  size_t i  = (size_t)blockIdx.x * 256 + threadIdx.x;    // over BB*WIN_*DD
  size_t d  = i & (DD - 1);
  size_t bw = i >> 11;
  size_t w  = bw & (WIN_ - 1);
  size_t b  = bw >> 7;
  wx[i] = (__bf16)x[((b * SS + WSs + w) << 11) + d];
}

// ---------------------------------------------------------------------------
// GEMM: Out[M,N] = A[M,K](bf16) * W[N,K](f32, converted to bf16 on stage-in)^T
// Block tile 128x128, 8 waves of 32(M)x64(N), K-stage 64 (2 WMMA k-steps per
// barrier pair). A-tiles staged via GLOBAL_LOAD_ASYNC_TO_LDS_B128 (ASYNCcnt);
// W-tiles go through VGPRs for the f32->bf16 convert.
// ---------------------------------------------------------------------------
template <bool OUT_BF16>
__global__ __launch_bounds__(256) void gemm_xwt_k(const __bf16* __restrict__ A,
                                                  const float*  __restrict__ W,
                                                  void* __restrict__ Out,
                                                  int M, int N, int K) {
  __shared__ __bf16 As[128 * 64];   // 16 KB
  __shared__ __bf16 Bs[128 * 64];   // 16 KB

  const int t     = threadIdx.x;
  const int wid   = t >> 5;
  const int lane  = t & 31;
  const int l16   = lane & 15;
  const int lhalf = lane >> 4;
  const int m0    = blockIdx.y * 128;
  const int n0    = blockIdx.x * 128;
  const int waveM = (wid & 3) * 32;
  const int waveN = (wid >> 2) * 64;

  const v8f vzero = {0.f, 0.f, 0.f, 0.f, 0.f, 0.f, 0.f, 0.f};
  v8f acc[2][4];
#pragma unroll
  for (int i = 0; i < 2; ++i)
#pragma unroll
    for (int j = 0; j < 4; ++j) acc[i][j] = vzero;

  const int kTiles = K >> 6;
  for (int kt = 0; kt < kTiles; ++kt) {
    const int k0 = kt << 6;
    // Stage A tile (128x64 bf16): 1024 16-byte chunks, 4 per thread, async
#pragma unroll
    for (int c = t; c < 1024; c += 256) {
      int row = c >> 3, col8 = (c & 7) << 3;
      async_copy_b128(A + (size_t)(m0 + row) * K + k0 + col8,
                      As + row * 64 + col8);
    }
    // Stage W tile (128x64 f32 -> bf16): one row-half (32 floats) per thread
    {
      int rn = t >> 1, ch = (t & 1) << 5;
      const float* wp = W + (size_t)(n0 + rn) * K + k0 + ch;
      union { v8bf v[4]; __bf16 e[32]; } bt;
#pragma unroll
      for (int i = 0; i < 32; i += 4) {
        v4f f = *(const v4f*)(wp + i);
        bt.e[i + 0] = (__bf16)f.x; bt.e[i + 1] = (__bf16)f.y;
        bt.e[i + 2] = (__bf16)f.z; bt.e[i + 3] = (__bf16)f.w;
      }
#pragma unroll
      for (int u = 0; u < 4; ++u)
        *(v8bf*)(Bs + rn * 64 + ch + u * 8) = bt.v[u];
    }
    if (kt + 1 < kTiles) {   // emits global_prefetch_b8 (CDNA5 prefetch path)
      __builtin_prefetch(A + (size_t)(m0 + (t & 127)) * K + k0 + 64, 0, 1);
      __builtin_prefetch(W + (size_t)(n0 + (t & 127)) * K + k0 + 64, 0, 1);
    }
    async_wait0();       // ASYNCcnt == 0: A tile landed in LDS
    __syncthreads();

#pragma unroll
    for (int kk = 0; kk < 2; ++kk) {
      Frag a[2];
#pragma unroll
      for (int i = 0; i < 2; ++i) {
        int row = waveM + i * 16 + l16;
        int klo = kk * 32 + lhalf * 8;
        a[i].h[0] = *(const v8bf*)(As + row * 64 + klo);
        a[i].h[1] = *(const v8bf*)(As + row * 64 + 16 + klo);
      }
#pragma unroll
      for (int j = 0; j < 4; ++j) {
        Frag b;
        int n   = waveN + j * 16 + l16;
        int kb0 = kk * 32 + lhalf * 8;
        b.h[0] = *(const v8bf*)(Bs + n * 64 + kb0);
        b.h[1] = *(const v8bf*)(Bs + n * 64 + kb0 + 16);
#pragma unroll
        for (int i = 0; i < 2; ++i) acc[i][j] = wmma_bf16(a[i], b, acc[i][j]);
      }
    }
    __syncthreads();
  }

  // Epilogue: D layout -> vgpr v holds row v + 8*(lane/16), col = lane%16
#pragma unroll
  for (int i = 0; i < 2; ++i)
#pragma unroll
    for (int j = 0; j < 4; ++j)
#pragma unroll
      for (int v = 0; v < 8; ++v) {
        int row = m0 + waveM + i * 16 + v + lhalf * 8;
        int col = n0 + waveN + j * 16 + l16;
        float val = acc[i][j][v];
        if (OUT_BF16) ((__bf16*)Out)[(size_t)row * N + col] = (__bf16)val;
        else          ((float*) Out)[(size_t)row * N + col] = val;
      }
}

// ---------------------------------------------------------------------------
// Attention per (query-tile of 128 chunks, head, batch).
// K tile staged async HBM->LDS; V staged through VGPRs (needs transpose).
// scores = qc*K^T (WMMA) -> mask -> softmax (register-resident, shfl within
// 16-lane halves) -> P (bf16, LDS, reusing K buffer) -> O = P*V (WMMA).
// ---------------------------------------------------------------------------
__global__ __launch_bounds__(256) void attn_k(const __bf16* __restrict__ qcb,
                                              const __bf16* __restrict__ kb,
                                              const __bf16* __restrict__ vb,
                                              const int*    __restrict__ amask,
                                              __bf16* __restrict__ attnb) {
  __shared__ __bf16 ldsK[128 * 128];   // K tile, later reused for P
  __shared__ __bf16 ldsVt[128 * 128];  // V^T tile: [dh][key]

  const int t     = threadIdx.x;
  const int wid   = t >> 5;
  const int lane  = t & 31;
  const int l16   = lane & 15;
  const int lhalf = lane >> 4;
  const int qt    = blockIdx.x;        // 0..7 (128 chunks each)
  const int h     = blockIdx.y;        // 0..15
  const int b     = blockIdx.z;        // 0..3

  const size_t kvBase = ((size_t)b * WIN_) * DD + (size_t)h * DHh;

  // Cooperative stage: K async row-major, V transposed (scatter) into LDS
#pragma unroll
  for (int c = t; c < 2048; c += 256) {
    int key = c >> 4, col8 = (c & 15) << 3;
    async_copy_b128(kb + kvBase + (size_t)key * DD + col8,
                    ldsK + key * 128 + col8);
    v8bf vvv = *(const v8bf*)(vb + kvBase + (size_t)key * DD + col8);
#pragma unroll
    for (int e = 0; e < 8; ++e) ldsVt[(col8 + e) * 128 + key] = vvv[e];
  }
  async_wait0();
  __syncthreads();

  const int c0 = qt * 128 + wid * 16;  // wave's 16 query chunks
  const v8f vzero = {0.f, 0.f, 0.f, 0.f, 0.f, 0.f, 0.f, 0.f};
  v8f acc[8];
#pragma unroll
  for (int j = 0; j < 8; ++j) acc[j] = vzero;

  // scores: 16(q) x 128(key), K-dim = dh = 128 (4 steps of 32)
  const size_t qrow = ((size_t)b * CSc + c0 + l16) * DD + (size_t)h * DHh;
#pragma unroll
  for (int kk = 0; kk < 4; ++kk) {
    Frag a;
    int klo = kk * 32 + lhalf * 8;
    a.h[0] = *(const v8bf*)(qcb + qrow + klo);
    a.h[1] = *(const v8bf*)(qcb + qrow + klo + 16);
#pragma unroll
    for (int j = 0; j < 8; ++j) {
      Frag bf;
      int n   = j * 16 + l16;
      int kb0 = kk * 32 + lhalf * 8;
      bf.h[0] = *(const v8bf*)(ldsK + n * 128 + kb0);
      bf.h[1] = *(const v8bf*)(ldsK + n * 128 + kb0 + 16);
      acc[j] = wmma_bf16(a, bf, acc[j]);
    }
  }

  // mask (causal-in-window | pre-window->key0, and pad mask), then scale
  const float scale = 0.0883883476483184f;  // 1/sqrt(128)
#pragma unroll
  for (int j = 0; j < 8; ++j) {
    int key  = j * 16 + l16;
    int kpos = WSs + key;
    int pad  = amask[(size_t)b * SS + kpos];
#pragma unroll
    for (int v = 0; v < 8; ++v) {
      int q    = c0 + v + lhalf * 8;
      int qpos = q * RR + (RR - 1);
      bool ok  = (pad != 0) && ((kpos <= qpos) || (key == 0 && qpos < WSs));
      acc[j][v] = ok ? acc[j][v] * scale : NEGV;
    }
  }

  // softmax per row: rows live as (vgpr v, lane-half); reduce across 16 lanes
#pragma unroll
  for (int v = 0; v < 8; ++v) {
    float m = -3.4e38f;
#pragma unroll
    for (int j = 0; j < 8; ++j) m = fmaxf(m, acc[j][v]);
#pragma unroll
    for (int off = 1; off < 16; off <<= 1) m = fmaxf(m, __shfl_xor(m, off, 16));
    float s = 0.f;
#pragma unroll
    for (int j = 0; j < 8; ++j) { float p = __expf(acc[j][v] - m); acc[j][v] = p; s += p; }
#pragma unroll
    for (int off = 1; off < 16; off <<= 1) s += __shfl_xor(s, off, 16);
    float inv = 1.f / s;
#pragma unroll
    for (int j = 0; j < 8; ++j) acc[j][v] *= inv;
  }

  __syncthreads();  // everyone finished reading ldsK (K tile)

  // write P (bf16) to wave-private LDS region (reusing ldsK)
  __bf16* P = ldsK + wid * 16 * 128;
#pragma unroll
  for (int j = 0; j < 8; ++j)
#pragma unroll
    for (int v = 0; v < 8; ++v)
      P[(v + lhalf * 8) * 128 + j * 16 + l16] = (__bf16)acc[j][v];
  __syncthreads();

  // O = P(16x128) * V(128x128): accumulate over keys, N = dh
  v8f accO[8];
#pragma unroll
  for (int j = 0; j < 8; ++j) accO[j] = vzero;
#pragma unroll
  for (int kk = 0; kk < 4; ++kk) {
    Frag a;
    int klo = kk * 32 + lhalf * 8;
    a.h[0] = *(const v8bf*)(P + l16 * 128 + klo);
    a.h[1] = *(const v8bf*)(P + l16 * 128 + klo + 16);
#pragma unroll
    for (int j = 0; j < 8; ++j) {
      Frag bf;
      int n   = j * 16 + l16;
      int kb0 = kk * 32 + lhalf * 8;
      bf.h[0] = *(const v8bf*)(ldsVt + n * 128 + kb0);
      bf.h[1] = *(const v8bf*)(ldsVt + n * 128 + kb0 + 16);
      accO[j] = wmma_bf16(a, bf, accO[j]);
    }
  }

  // store O (bf16) into attn output (B*Cs, D) at head columns
#pragma unroll
  for (int j = 0; j < 8; ++j)
#pragma unroll
    for (int v = 0; v < 8; ++v) {
      int q = c0 + v + lhalf * 8;
      attnb[((size_t)b * CSc + q) * DD + h * DHh + j * 16 + l16] = (__bf16)accO[j][v];
    }
}

// ---------------------------------------------------------------------------
// Linear resize rows 1024 -> 4096 (half-pixel centers, edge clamp), f32 out
// ---------------------------------------------------------------------------
__global__ __launch_bounds__(256) void resize_k(const float* __restrict__ oc,
                                                float* __restrict__ out) {
  size_t i  = (size_t)blockIdx.x * 256 + threadIdx.x;   // over BB*SS*(DD/4)
  size_t d4 = i & 511;
  size_t bs = i >> 9;
  size_t s  = bs & (SS - 1);
  size_t b  = bs >> 12;
  float coord = ((float)s + 0.5f) * 0.25f - 0.5f;
  float f  = floorf(coord);
  float w  = coord - f;
  int   i0 = (int)f;
  int   i1 = i0 + 1;
  i0 = i0 < 0 ? 0 : (i0 > CSc - 1 ? CSc - 1 : i0);
  i1 = i1 < 0 ? 0 : (i1 > CSc - 1 ? CSc - 1 : i1);
  v4f a = *((const v4f*)(oc + (((size_t)b * CSc + i0) << 11)) + d4);
  v4f c = *((const v4f*)(oc + (((size_t)b * CSc + i1) << 11)) + d4);
  v4f r;
  r.x = a.x + (c.x - a.x) * w;
  r.y = a.y + (c.y - a.y) * w;
  r.z = a.z + (c.z - a.z) * w;
  r.w = a.w + (c.w - a.w) * w;
  *((v4f*)(out + (((size_t)b * SS + s) << 11)) + d4) = r;
}

// ---------------------------------------------------------------------------
extern "C" void kernel_launch(void* const* d_in, const int* in_sizes, int n_in,
                              void* d_out, int out_size, void* d_ws, size_t ws_size,
                              hipStream_t stream) {
  (void)in_sizes; (void)n_in; (void)out_size;
  const float* x     = (const float*)d_in[0];
  const int*   amask = (const int*)  d_in[1];
  const float* Wq    = (const float*)d_in[2];
  const float* Wk    = (const float*)d_in[3];
  const float* Wv    = (const float*)d_in[4];
  const float* Wo    = (const float*)d_in[5];

  // Workspace layout (bytes)
  const size_t SZ_XC   = (size_t)BB * CSc * DD * 2;   // 16 MB
  const size_t SZ_WX   = (size_t)BB * WIN_ * DD * 2;  //  2 MB
  const size_t SZ_QC   = SZ_XC;                       // 16 MB
  const size_t SZ_KV   = SZ_WX;                       //  2 MB each
  const size_t SZ_ATT  = SZ_XC;                       // 16 MB
  const size_t SZ_OUTC = (size_t)BB * CSc * DD * 4;   // 32 MB
  const size_t NEED = SZ_XC + SZ_WX + SZ_QC + 2 * SZ_KV + SZ_ATT + SZ_OUTC;
  if (ws_size < NEED) return;

  char* p = (char*)d_ws;
  __bf16* xcb   = (__bf16*)p; p += SZ_XC;
  __bf16* wxb   = (__bf16*)p; p += SZ_WX;
  __bf16* qcb   = (__bf16*)p; p += SZ_QC;
  __bf16* kbuf  = (__bf16*)p; p += SZ_KV;
  __bf16* vbuf  = (__bf16*)p; p += SZ_KV;
  __bf16* attnb = (__bf16*)p; p += SZ_ATT;
  float*  outc  = (float*)p;

  // 1) chunk-mean (linearity: mean(X)@Wq^T == mean(X@Wq^T)) + window slice
  prep_mean_k<<<(BB * CSc * DD) / 256, 256, 0, stream>>>(x, xcb);
  prep_win_k <<<(BB * WIN_ * DD) / 256, 256, 0, stream>>>(x, wxb);

  // 2) projections (WMMA bf16)
  gemm_xwt_k<true ><<<dim3(16, 32), 256, 0, stream>>>(xcb, Wq, qcb,  BB * CSc, DD, DD);
  gemm_xwt_k<true ><<<dim3(16,  4), 256, 0, stream>>>(wxb, Wk, kbuf, BB * WIN_, DD, DD);
  gemm_xwt_k<true ><<<dim3(16,  4), 256, 0, stream>>>(wxb, Wv, vbuf, BB * WIN_, DD, DD);

  // 3) attention (WMMA scores + softmax + WMMA PV)
  attn_k<<<dim3(CSc / 128, HH, BB), 256, 0, stream>>>(qcb, kbuf, vbuf, amask, attnb);

  // 4) output projection (f32 result)
  gemm_xwt_k<false><<<dim3(16, 32), 256, 0, stream>>>(attnb, Wo, outc, BB * CSc, DD, DD);

  // 5) linear resize 1024 -> 4096 rows
  resize_k<<<(BB * SS * (DD / 4)) / 256, 256, 0, stream>>>(outc, (float*)d_out);
}
